// TransformerEncoder_3315714752641
// MI455X (gfx1250) — matrix operand order repaired and checked by reading, not compiled, not run on hardware
//
#include <hip/hip_runtime.h>
#include <hip/hip_bf16.h>
#include <cstdint>

// ---------------- problem constants ----------------
#define Bn   16
#define Ln   512
#define Dn   512
#define Hn   16
#define Sn   3
#define FFn  2048
#define DHn  32          // Dn / Hn
#define NHn  256         // Hn * Bn
#define Mn   8192        // Bn * Ln
#define EPSn 1e-5f
#define BN_CHUNKS 64

typedef __attribute__((ext_vector_type(16))) _Float16 v16h;
typedef __attribute__((ext_vector_type(8)))  float    v8f;
typedef __attribute__((ext_vector_type(2)))  float    v2f;

union Frag16 { v16h v; struct { uint4 lo, hi; } q; };

// ---------------- WMMA fragment loaders (CDNA5 wave32 layouts) ----------------
// A 16x32 f16 (MxK): lane row = lane&15; lanes<16 hold K {0..7,16..23}, lanes>=16 hold {8..15,24..31}.
// Both halves are 16B-contiguous -> two b128 loads.
__device__ __forceinline__ v16h load_a_frag(const _Float16* A, int lda, int row0, int k0) {
    const int lane = threadIdx.x & 31;
    const _Float16* p = A + (size_t)(row0 + (lane & 15)) * lda + k0 + ((lane < 16) ? 0 : 8);
    Frag16 f;
    f.q.lo = *reinterpret_cast<const uint4*>(p);
    f.q.hi = *reinterpret_cast<const uint4*>(p + 16);
    return f.v;
}

// B 32x16 f16 (KxN) sourced from a TRANSPOSED buffer Bt (N rows, K cols):
// lane col = lane&15; lanes<16 hold K {0..7,16..23}, lanes>=16 hold {8..15,24..31} -> two b128 loads.
__device__ __forceinline__ v16h load_bT_frag(const _Float16* Bt, int ldb, int k0, int col0) {
    const int lane = threadIdx.x & 31;
    const _Float16* p = Bt + (size_t)(col0 + (lane & 15)) * ldb + k0 + ((lane < 16) ? 0 : 8);
    Frag16 f;
    f.q.lo = *reinterpret_cast<const uint4*>(p);
    f.q.hi = *reinterpret_cast<const uint4*>(p + 16);
    return f.v;
}

#define WMMA_F16(a, b, c) \
    __builtin_amdgcn_wmma_f32_16x16x32_f16(false, (a), false, (b), (short)0, (c), false, false)

// ---------------- converters ----------------
__global__ void f32_to_f16_kernel(const float* __restrict__ src, _Float16* __restrict__ dst, size_t n) {
    size_t i = (size_t)blockIdx.x * blockDim.x + threadIdx.x;
    if (i < n) dst[i] = (_Float16)src[i];
}

// W (K,N) fp32 -> Wt (N,K) f16, LDS-tiled 32x32 transpose. block = 256 (32x8), grid = (K/32, N/32).
__global__ void transpose_cvt(const float* __restrict__ W, _Float16* __restrict__ Wt, int K, int N) {
    __shared__ float tile[32][33];
    const int k0 = blockIdx.x * 32;
    const int n0 = blockIdx.y * 32;
    const int tx = threadIdx.x & 31;
    const int ty = threadIdx.x >> 5;
#pragma unroll
    for (int j = 0; j < 32; j += 8)
        tile[ty + j][tx] = W[(size_t)(k0 + ty + j) * N + (n0 + tx)];
    __syncthreads();
#pragma unroll
    for (int j = 0; j < 32; j += 8)
        Wt[(size_t)(n0 + ty + j) * K + (k0 + tx)] = (_Float16)tile[tx][ty + j];
}

// V (B*L, D) f16 -> Vt (B,H,DH,L) f16 (per-plane 32x32 transpose). grid = (L/32, B*H).
__global__ void transpose_v(const _Float16* __restrict__ vh, _Float16* __restrict__ vt) {
    __shared__ _Float16 tile[32][34];
    const int l0 = blockIdx.x * 32;
    const int bh = blockIdx.y;         // plane index b*Hn + h
    const int b  = bh / Hn;
    const int h  = bh % Hn;
    const int tx = threadIdx.x & 31;
    const int ty = threadIdx.x >> 5;
    const _Float16* src = vh + (size_t)b * Ln * Dn + h * DHn;
#pragma unroll
    for (int j = 0; j < 32; j += 8)
        tile[ty + j][tx] = src[(size_t)(l0 + ty + j) * Dn + tx];   // tile[l][d]
    __syncthreads();
    _Float16* dst = vt + (size_t)bh * DHn * Ln;                    // (DH, L) plane
#pragma unroll
    for (int j = 0; j < 32; j += 8)
        dst[(size_t)(ty + j) * Ln + (l0 + tx)] = tile[tx][ty + j];
}

// ---------------- generic WMMA GEMM ----------------
// C = act(A(MxK,f16) @ Bt^T + bias) [+resid]; Bt is (N,K) f16. Each wave: 16x64 strip (4 WMMA/k-step).
// block = 256 (8 waves), grid = (N/64, M/128).
template <bool RELU, bool BIAS, bool OUTF, bool OUTH, bool RESID>
__global__ void gemm_wmma(const _Float16* __restrict__ A, const _Float16* __restrict__ Bt,
                          const float* __restrict__ bias,
                          float* __restrict__ Cf, _Float16* __restrict__ Ch,
                          const float* __restrict__ resid,
                          int M, int N, int K) {
    const int wave = threadIdx.x >> 5;
    const int n0   = blockIdx.x * 64;
    const int row0 = (blockIdx.y * 8 + wave) * 16;

    v8f acc[4] = {v8f{}, v8f{}, v8f{}, v8f{}};
    for (int k0 = 0; k0 < K; k0 += 32) {
        if (k0 + 32 < K) {  // gfx1250 global_prefetch_b8 for the next k-step
            __builtin_prefetch((const void*)(A + (size_t)(row0 + (threadIdx.x & 15)) * K + k0 + 32), 0, 3);
            __builtin_prefetch((const void*)(Bt + (size_t)(n0 + (threadIdx.x & 31)) * K + k0 + 32), 0, 3);
        }
        v16h a  = load_a_frag(A, K, row0, k0);
        v16h b0 = load_bT_frag(Bt, K, k0, n0);
        v16h b1 = load_bT_frag(Bt, K, k0, n0 + 16);
        v16h b2 = load_bT_frag(Bt, K, k0, n0 + 32);
        v16h b3 = load_bT_frag(Bt, K, k0, n0 + 48);
        acc[0] = WMMA_F16(a, b0, acc[0]);
        acc[1] = WMMA_F16(a, b1, acc[1]);
        acc[2] = WMMA_F16(a, b2, acc[2]);
        acc[3] = WMMA_F16(a, b3, acc[3]);
    }

    const int lane  = threadIdx.x & 31;
    const int rbase = row0 + ((lane < 16) ? 0 : 8);
#pragma unroll
    for (int t = 0; t < 4; ++t) {
        const int col = n0 + t * 16 + (lane & 15);
        const float bv = BIAS ? bias[col] : 0.0f;
#pragma unroll
        for (int r = 0; r < 8; ++r) {
            float v = acc[t][r] + bv;
            if (RELU) v = fmaxf(v, 0.0f);
            const size_t idx = (size_t)(rbase + r) * N + col;
            if (RESID) v += resid[idx];
            if (OUTF) Cf[idx] = v;
            if (OUTH) Ch[idx] = (_Float16)v;
        }
    }
}

// ---------------- attention scores: sc[n,q,k] = (Q_ @ K_^T)/sqrt(DH) ----------------
// DH==32 -> one k-step; each wave does 4 k-tiles per Q fragment. grid = (L/64, (L/16)/8, NH).
__global__ void attn_scores(const _Float16* __restrict__ Qh, const _Float16* __restrict__ Kh,
                            float* __restrict__ sc) {
    const int wave = threadIdx.x >> 5;
    const int kt0  = blockIdx.x * 64;
    const int qt   = blockIdx.y * 8 + wave;
    const int n    = blockIdx.z;
    const int h    = n / Bn;
    const int b    = n % Bn;

    const _Float16* Qn = Qh + (size_t)b * Ln * Dn + h * DHn;   // (L,DH) view, ld=Dn
    const _Float16* Kn = Kh + (size_t)b * Ln * Dn + h * DHn;

    v16h a = load_a_frag(Qn, Dn, qt * 16, 0);
    v8f acc[4];
#pragma unroll
    for (int t = 0; t < 4; ++t) {
        v16h bf = load_bT_frag(Kn, Dn, 0, kt0 + t * 16);
        v8f z = {};
        acc[t] = WMMA_F16(a, bf, z);
    }

    const float scale = 0.17677669529663687f;                  // 1/sqrt(32)
    const int lane = threadIdx.x & 31;
    const int r0   = qt * 16 + ((lane < 16) ? 0 : 8);
    float* dst = sc + (size_t)n * Ln * Ln;
#pragma unroll
    for (int t = 0; t < 4; ++t) {
        const int col = kt0 + t * 16 + (lane & 15);
#pragma unroll
        for (int r = 0; r < 8; ++r)
            dst[(size_t)(r0 + r) * Ln + col] = acc[t][r] * scale;
    }
}

// ---------------- softmax over n = H*B axis (float2 vectorized) ----------------
__global__ void softmax_n(const float* __restrict__ sc, _Float16* __restrict__ att) {
    const size_t idx2 = ((size_t)blockIdx.x * blockDim.x + threadIdx.x) * 2;
    if (idx2 >= (size_t)Ln * Ln) return;
    const size_t stride = (size_t)Ln * Ln;
    float m0 = -3.4e38f, m1 = -3.4e38f;
    for (int n = 0; n < NHn; ++n) {
        v2f v = *reinterpret_cast<const v2f*>(sc + (size_t)n * stride + idx2);
        m0 = fmaxf(m0, v[0]); m1 = fmaxf(m1, v[1]);
    }
    float s0 = 0.0f, s1 = 0.0f;
    for (int n = 0; n < NHn; ++n) {
        v2f v = *reinterpret_cast<const v2f*>(sc + (size_t)n * stride + idx2);
        s0 += __expf(v[0] - m0); s1 += __expf(v[1] - m1);
    }
    const float i0 = 1.0f / s0, i1 = 1.0f / s1;
    for (int n = 0; n < NHn; ++n) {
        v2f v = *reinterpret_cast<const v2f*>(sc + (size_t)n * stride + idx2);
        union { _Float16 h[2]; uint32_t u; } pk;
        pk.h[0] = (_Float16)(__expf(v[0] - m0) * i0);
        pk.h[1] = (_Float16)(__expf(v[1] - m1) * i1);
        *reinterpret_cast<uint32_t*>(att + (size_t)n * stride + idx2) = pk.u;
    }
}

// ---------------- attention output: o = att @ V_, merged heads + residual ----------------
// Each wave: 16 q-rows x 32 d-cols (2 WMMA per att fragment). grid = ((L/16)/8, NH).
__global__ void attn_out(const _Float16* __restrict__ att, const _Float16* __restrict__ Vt,
                         const float* __restrict__ xin, float* __restrict__ t) {
    const int wave = threadIdx.x >> 5;
    const int qt   = blockIdx.x * 8 + wave;
    const int n    = blockIdx.y;
    const int h    = n / Bn;
    const int b    = n % Bn;

    const _Float16* An  = att + (size_t)n * Ln * Ln;               // (L,L), ld=Ln
    const _Float16* Vtn = Vt + (size_t)(b * Hn + h) * DHn * Ln;    // (DH,L), ld=Ln

    v8f acc0 = {}, acc1 = {};
    for (int k0 = 0; k0 < Ln; k0 += 32) {
        v16h a  = load_a_frag(An, Ln, qt * 16, k0);
        v16h b0 = load_bT_frag(Vtn, Ln, k0, 0);
        v16h b1 = load_bT_frag(Vtn, Ln, k0, 16);
        acc0 = WMMA_F16(a, b0, acc0);
        acc1 = WMMA_F16(a, b1, acc1);
    }

    const int lane = threadIdx.x & 31;
    const int r0   = qt * 16 + ((lane < 16) ? 0 : 8);
    const int c0   = h * DHn + (lane & 15);
#pragma unroll
    for (int r = 0; r < 8; ++r) {
        size_t i0 = (size_t)b * Ln * Dn + (size_t)(r0 + r) * Dn + c0;
        t[i0]      = acc0[r] + xin[i0];
        t[i0 + 16] = acc1[r] + xin[i0 + 16];
    }
}

// ---------------- BatchNorm: two-stage row-coalesced reduction ----------------
__global__ void bn_partial(const float* __restrict__ X, float* __restrict__ psum,
                           float* __restrict__ psq) {
    const int blk = blockIdx.x;                 // BN_CHUNKS blocks
    const int tid = threadIdx.x;                // 256 threads; 2 channels each
    float s0 = 0, s1 = 0, q0 = 0, q1 = 0;
    const int rows = Mn / BN_CHUNKS;
    const int r0 = blk * rows;
    for (int r = r0; r < r0 + rows; ++r) {
        const float* row = X + (size_t)r * Dn;
        float v0 = row[tid], v1 = row[tid + 256];
        s0 += v0; q0 += v0 * v0;
        s1 += v1; q1 += v1 * v1;
    }
    psum[(size_t)blk * Dn + tid]       = s0;
    psum[(size_t)blk * Dn + tid + 256] = s1;
    psq [(size_t)blk * Dn + tid]       = q0;
    psq [(size_t)blk * Dn + tid + 256] = q1;
}

__global__ void bn_finalize(const float* __restrict__ psum, const float* __restrict__ psq,
                            float* __restrict__ mean, float* __restrict__ rstd) {
    const int ch = blockIdx.x * blockDim.x + threadIdx.x;
    if (ch >= Dn) return;
    float s = 0, q = 0;
    for (int i = 0; i < BN_CHUNKS; ++i) { s += psum[(size_t)i * Dn + ch]; q += psq[(size_t)i * Dn + ch]; }
    const float m = s / (float)Mn;
    mean[ch] = m;
    rstd[ch] = rsqrtf(q / (float)Mn - m * m + EPSn);
}

__global__ void bn_apply(const float* __restrict__ X, const float* __restrict__ mean,
                         const float* __restrict__ rstd, const float* __restrict__ g,
                         const float* __restrict__ bt, float* __restrict__ Yf,
                         _Float16* __restrict__ Yh) {
    size_t i = (size_t)blockIdx.x * blockDim.x + threadIdx.x;
    if (i >= (size_t)Mn * Dn) return;
    const int ch = (int)(i & (Dn - 1));
    const float v = (X[i] - mean[ch]) * rstd[ch] * g[ch] + bt[ch];
    Yf[i] = v;
    Yh[i] = (_Float16)v;
}

// ---------------- host orchestration ----------------
extern "C" void kernel_launch(void* const* d_in, const int* in_sizes, int n_in,
                              void* d_out, int out_size, void* d_ws, size_t ws_size,
                              hipStream_t stream) {
    const float* x     = (const float*)d_in[0];
    const float* emb   = (const float*)d_in[1];
    const float* bn2_g = (const float*)d_in[2];
    const float* bn2_b = (const float*)d_in[3];
    const float* Wq    = (const float*)d_in[4];
    const float* bq    = (const float*)d_in[5];
    const float* Wk    = (const float*)d_in[6];
    const float* bk    = (const float*)d_in[7];
    const float* Wv    = (const float*)d_in[8];
    const float* bv    = (const float*)d_in[9];
    const float* bna_g = (const float*)d_in[10];
    const float* bna_b = (const float*)d_in[11];
    const float* W1    = (const float*)d_in[12];
    const float* b1    = (const float*)d_in[13];
    const float* W2    = (const float*)d_in[14];
    const float* b2    = (const float*)d_in[15];
    const float* bnf_g = (const float*)d_in[16];
    const float* bnf_b = (const float*)d_in[17];
    float* out = (float*)d_out;
    (void)in_sizes; (void)n_in; (void)out_size; (void)ws_size;

    char* base = (char*)d_ws;
    size_t off = 0;
    auto alloc = [&](size_t bytes) -> void* {
        off = (off + 255) & ~(size_t)255;
        void* p = base + off;
        off += bytes;
        return p;
    };
    _Float16* x_h   = (_Float16*)alloc((size_t)Mn * Dn * 2);
    _Float16* emb_t = (_Float16*)alloc((size_t)Dn * Dn * 2);        // (out,in)
    _Float16* wq_t  = (_Float16*)alloc((size_t)Sn * Dn * Dn * 2);   // (N=D, K=D) per layer
    _Float16* wk_t  = (_Float16*)alloc((size_t)Sn * Dn * Dn * 2);
    _Float16* wv_t  = (_Float16*)alloc((size_t)Sn * Dn * Dn * 2);
    _Float16* w1_t  = (_Float16*)alloc((size_t)Sn * FFn * Dn * 2);  // (N=FF, K=D)
    _Float16* w2_t  = (_Float16*)alloc((size_t)Sn * Dn * FFn * 2);  // (N=D, K=FF)
    float*    enc_f = (float*)   alloc((size_t)Mn * Dn * 4);
    _Float16* enc_h = (_Float16*)alloc((size_t)Mn * Dn * 2);
    float*    t_f   = (float*)   alloc((size_t)Mn * Dn * 4);
    _Float16* qh    = (_Float16*)alloc((size_t)Mn * Dn * 2);
    _Float16* kh    = (_Float16*)alloc((size_t)Mn * Dn * 2);
    _Float16* vh    = (_Float16*)alloc((size_t)Mn * Dn * 2);
    _Float16* vt    = (_Float16*)alloc((size_t)Mn * Dn * 2);        // (B,H,DH,L)
    _Float16* h1_h  = (_Float16*)alloc((size_t)Mn * FFn * 2);
    float*    sc_f  = (float*)   alloc((size_t)NHn * Ln * Ln * 4);
    _Float16* att_h = (_Float16*)alloc((size_t)NHn * Ln * Ln * 2);
    float*    psum  = (float*)   alloc((size_t)BN_CHUNKS * Dn * 4);
    float*    psq   = (float*)   alloc((size_t)BN_CHUNKS * Dn * 4);
    float*    mean  = (float*)   alloc((size_t)Dn * 4);
    float*    rstd  = (float*)   alloc((size_t)Dn * 4);

    const dim3 blk(256);

    // ---- weight preprocessing: fp32 (K,N) -> f16 (N,K) ----
    f32_to_f16_kernel<<<(unsigned)(((size_t)Mn * Dn + 255) / 256), blk, 0, stream>>>(x, x_h, (size_t)Mn * Dn);
    transpose_cvt<<<dim3(Dn / 32, Dn / 32), blk, 0, stream>>>(emb, emb_t, Dn, Dn);
    for (int s = 0; s < Sn; ++s) {
        transpose_cvt<<<dim3(Dn / 32, Dn / 32), blk, 0, stream>>>(Wq + (size_t)s * Dn * Dn, wq_t + (size_t)s * Dn * Dn, Dn, Dn);
        transpose_cvt<<<dim3(Dn / 32, Dn / 32), blk, 0, stream>>>(Wk + (size_t)s * Dn * Dn, wk_t + (size_t)s * Dn * Dn, Dn, Dn);
        transpose_cvt<<<dim3(Dn / 32, Dn / 32), blk, 0, stream>>>(Wv + (size_t)s * Dn * Dn, wv_t + (size_t)s * Dn * Dn, Dn, Dn);
        transpose_cvt<<<dim3(Dn / 32, FFn / 32), blk, 0, stream>>>(W1 + (size_t)s * Dn * FFn, w1_t + (size_t)s * FFn * Dn, Dn, FFn);
        transpose_cvt<<<dim3(FFn / 32, Dn / 32), blk, 0, stream>>>(W2 + (size_t)s * FFn * Dn, w2_t + (size_t)s * Dn * FFn, FFn, Dn);
    }

    const unsigned gy        = Mn / 128;     // 8 waves x 16 rows per block
    const unsigned eltBlocks = (unsigned)(((size_t)Mn * Dn + 255) / 256);
    auto run_bn = [&](const float* X, const float* g, const float* b2_, float* Yf, _Float16* Yh) {
        bn_partial<<<BN_CHUNKS, blk, 0, stream>>>(X, psum, psq);
        bn_finalize<<<Dn / 256, blk, 0, stream>>>(psum, psq, mean, rstd);
        bn_apply<<<eltBlocks, blk, 0, stream>>>(X, mean, rstd, g, b2_, Yf, Yh);
    };

    // ---- embedding 1x1 conv + BN ----
    gemm_wmma<false, false, true, false, false><<<dim3(Dn / 64, gy), blk, 0, stream>>>(
        x_h, emb_t, nullptr, t_f, nullptr, nullptr, Mn, Dn, Dn);
    run_bn(t_f, bn2_g, bn2_b, enc_f, enc_h);

    for (int i = 0; i < Sn; ++i) {
        // ---- Q/K/V projections (bias + ReLU, f16 out) ----
        gemm_wmma<true, true, false, true, false><<<dim3(Dn / 64, gy), blk, 0, stream>>>(
            enc_h, wq_t + (size_t)i * Dn * Dn, bq + i * Dn, nullptr, qh, nullptr, Mn, Dn, Dn);
        gemm_wmma<true, true, false, true, false><<<dim3(Dn / 64, gy), blk, 0, stream>>>(
            enc_h, wk_t + (size_t)i * Dn * Dn, bk + i * Dn, nullptr, kh, nullptr, Mn, Dn, Dn);
        gemm_wmma<true, true, false, true, false><<<dim3(Dn / 64, gy), blk, 0, stream>>>(
            enc_h, wv_t + (size_t)i * Dn * Dn, bv + i * Dn, nullptr, vh, nullptr, Mn, Dn, Dn);
        transpose_v<<<dim3(Ln / 32, Bn * Hn), blk, 0, stream>>>(vh, vt);

        // ---- scores, softmax over n, att@V (fused residual into t_f) ----
        attn_scores<<<dim3(Ln / 64, (Ln / 16) / 8, NHn), blk, 0, stream>>>(qh, kh, sc_f);
        softmax_n<<<(unsigned)(((size_t)Ln * Ln / 2 + 255) / 256), blk, 0, stream>>>(sc_f, att_h);
        attn_out<<<dim3((Ln / 16) / 8, NHn), blk, 0, stream>>>(att_h, vt, enc_f, t_f);
        run_bn(t_f, bna_g + i * Dn, bna_b + i * Dn, enc_f, enc_h);

        // ---- FeedForward + residual + BN ----
        gemm_wmma<true, true, false, true, false><<<dim3(FFn / 64, gy), blk, 0, stream>>>(
            enc_h, w1_t + (size_t)i * FFn * Dn, b1 + i * FFn, nullptr, h1_h, nullptr, Mn, FFn, Dn);
        gemm_wmma<false, true, true, false, true><<<dim3(Dn / 64, gy), blk, 0, stream>>>(
            h1_h, w2_t + (size_t)i * Dn * FFn, b2 + i * Dn, t_f, nullptr, enc_f, Mn, Dn, FFn);
        float* yf = (i == Sn - 1) ? out : enc_f;
        run_bn(t_f, bnf_g + i * Dn, bnf_b + i * Dn, yf, enc_h);
    }
}
